// OptimizedMOEImproved_65180423685433
// MI455X (gfx1250) — matrix operand order repaired and checked by reading, not compiled, not run on hardware
//
#include <hip/hip_runtime.h>
#include <math.h>

// ---------------- problem constants ----------------
#define B_   16
#define C_   256
#define O_   256
#define Hh   56
#define Ww   56
#define HW   3136          // 56*56
#define E_   8
#define HID  512
#define K1   (C_ * 9)      // 2304; K reordered as k = r*256 + c (r = kh*3+kw)
#define NT   64            // pixel tile (N tile)
#define NTILES (HW / NT)   // 49
#define INV_SQRT_1PEPS 0.9999950000374997f  // 1/sqrt(1+1e-5)

typedef _Float16 v16h __attribute__((ext_vector_type(16)));
typedef _Float16 v8h  __attribute__((ext_vector_type(8)));
typedef float    v8f  __attribute__((ext_vector_type(8)));

static __device__ __forceinline__ v16h join8(v8h lo, v8h hi) {
  v16h r;
#pragma unroll
  for (int i = 0; i < 8; ++i) { r[i] = lo[i]; r[i + 8] = hi[i]; }
  return r;
}

static __device__ __forceinline__ v8f wmma_f16(v16h a, v16h b, v8f c) {
  // D = A(16x32 f16) * B(32x16 f16) + C(16x16 f32)
  return __builtin_amdgcn_wmma_f32_16x16x32_f16(
      /*neg_a=*/false, a, /*neg_b=*/false, b,
      /*c_mod=*/(short)0, c, /*reuse_a=*/false, /*reuse_b=*/false);
}

// fast SiLU: v * rcp(1+exp(-v)) -> v_exp_f32 + v_rcp_f32, no IEEE div chain
static __device__ __forceinline__ float silu_f(float v) {
  return v * __builtin_amdgcn_rcpf(1.0f + __expf(-v));
}

// ---------------- fp32 -> f16 convert (plain) ----------------
__global__ void k_f32_to_f16(const float* __restrict__ src,
                             _Float16* __restrict__ dst, int n) {
  int i = blockIdx.x * blockDim.x + threadIdx.x;
  int stride = gridDim.x * blockDim.x;
  for (; i < n; i += stride) dst[i] = (_Float16)src[i];
}

// ------- w1 convert + K-reorder: dst[e][m][r*256+c] = w1[e][m][c][kh][kw] ----
__global__ void k_convert_w1(const float* __restrict__ src,
                             _Float16* __restrict__ dst, int n) {
  int i = blockIdx.x * blockDim.x + threadIdx.x;
  int stride = gridDim.x * blockDim.x;
  for (; i < n; i += stride) {
    int em = i / K1;          // e*HID + m
    int t  = i % K1;
    int r  = t >> 8;          // 0..8  (kh*3+kw)
    int c  = t & 255;
    dst[i] = (_Float16)src[(em * C_ + c) * 9 + r];
  }
}

// ---------------- global average pool: one block per (b,c) ----------------
__global__ __launch_bounds__(256) void k_pool(const float* __restrict__ x,
                                              float* __restrict__ pooled) {
  __shared__ float red[256];
  int bc = blockIdx.x;
  const float* p = x + (size_t)bc * HW;
  float s = 0.0f;
  for (int i = threadIdx.x; i < HW; i += 256) s += p[i];
  red[threadIdx.x] = s;
  __syncthreads();
  for (int off = 128; off > 0; off >>= 1) {
    if (threadIdx.x < off) red[threadIdx.x] += red[threadIdx.x + off];
    __syncthreads();
  }
  if (threadIdx.x == 0) pooled[bc] = red[0] * (1.0f / (float)HW);
}

// ---------------- router: logits -> softmax -> top2 renorm -> dense gates ----
__global__ __launch_bounds__(128) void k_router(const float* __restrict__ pooled,
                                                const float* __restrict__ rw,
                                                const float* __restrict__ rb,
                                                float* __restrict__ gates) {
  __shared__ float lg[B_][E_];
  int t = threadIdx.x;
  if (t < B_ * E_) {
    int b = t / E_, e = t % E_;
    float s = rb[e];
    const float* pp = pooled + b * C_;
    const float* wp = rw + e * C_;
    for (int c = 0; c < C_; ++c) s += pp[c] * wp[c];
    lg[b][e] = s;
  }
  __syncthreads();
  if (t < B_) {
    float m = -1e30f;
    for (int e = 0; e < E_; ++e) m = fmaxf(m, lg[t][e]);
    float pr[E_]; float sum = 0.0f;
    for (int e = 0; e < E_; ++e) { pr[e] = __expf(lg[t][e] - m); sum += pr[e]; }
    float inv = 1.0f / sum;
    int i1 = 0, i2 = 0; float v1 = -1.0f, v2 = -1.0f;
    for (int e = 0; e < E_; ++e) {
      float p = pr[e] * inv;
      if (p > v1)      { v2 = v1; i2 = i1; v1 = p; i1 = e; }
      else if (p > v2) { v2 = p; i2 = e; }
    }
    float wsum = v1 + v2;
    for (int e = 0; e < E_; ++e) gates[t * E_ + e] = 0.0f;
    gates[t * E_ + i1] = v1 / wsum;
    gates[t * E_ + i2] = v2 / wsum;
  }
}

// ------- shared expert (1x1 conv, WMMA) + residual: writes d_out -------
__global__ __launch_bounds__(256) void k_shared(const _Float16* __restrict__ xh,
                                                const _Float16* __restrict__ swh,
                                                const float* __restrict__ scale,
                                                const float* __restrict__ bias,
                                                const float* __restrict__ xf,
                                                float* __restrict__ out) {
  __shared__ _Float16 xs[NT * C_];  // [n][c], k-contiguous, 32 KB
  int b = blockIdx.y;
  int n0 = blockIdx.x * NT;
  int tid = threadIdx.x;
  // vectorized stage: 16B global loads, scatter to n-major LDS
  for (int q = tid; q < (NT / 8) * C_; q += 256) {
    int c  = q >> 3;
    int i8 = q & 7;
    v8h v = *(const v8h*)&xh[(size_t)(b * C_ + c) * HW + n0 + i8 * 8];
#pragma unroll
    for (int u = 0; u < 8; ++u) xs[(i8 * 8 + u) * C_ + c] = v[u];
  }
  __syncthreads();

  int wave = tid >> 5, lane = tid & 31;
  int g = lane >> 4, ln = lane & 15;
  v8f acc[2][4] = {};
#pragma unroll 1
  for (int kt = 0; kt < C_ / 32; ++kt) {
    int k0 = kt * 32;
    v16h bfrag[4];
#pragma unroll
    for (int nt = 0; nt < 4; ++nt) {
      const _Float16* bp = &xs[(nt * 16 + ln) * C_ + k0 + g * 8];
      bfrag[nt] = join8(*(const v8h*)bp, *(const v8h*)(bp + 16));
    }
#pragma unroll
    for (int mc = 0; mc < 2; ++mc) {
      int m = (wave + mc * 8) * 16 + ln;
      const _Float16* ap = swh + (size_t)m * C_ + k0;
      v16h af = join8(*(const v8h*)(ap + g * 8), *(const v8h*)(ap + 16 + g * 8));
#pragma unroll
      for (int nt = 0; nt < 4; ++nt)
        acc[mc][nt] = wmma_f16(af, bfrag[nt], acc[mc][nt]);
    }
  }
#pragma unroll
  for (int mc = 0; mc < 2; ++mc)
#pragma unroll
    for (int j = 0; j < 8; ++j) {
      int m = (wave + mc * 8) * 16 + g * 8 + j;
      float inv = scale[m] * INV_SQRT_1PEPS;
      float bi = bias[m];
#pragma unroll
      for (int nt = 0; nt < 4; ++nt) {
        int n = n0 + nt * 16 + ln;
        size_t idx = ((size_t)(b * O_ + m)) * HW + n;
        float v = acc[mc][nt][j] * inv + bi;
        out[idx] = silu_f(v) + xf[idx];
      }
    }
}

// ------- fused routed expert: conv3x3+BN+SiLU (stage1) -> conv1x1+BN (stage2)
// one block per (pixel-tile, b*E+e); uniform early exit on gate==0.
// K order of stage1 is r-major (k = r*256 + c) matching k_convert_w1, so each
// 32-wide K-chunk has uniform (kh,kw) and contiguous channels: staging is
// divide-free and coalesced. Bs is double-buffered -> 1 barrier per K-step.
__global__ __launch_bounds__(256) void k_expert(const _Float16* __restrict__ xh,
                                                const _Float16* __restrict__ w1h,
                                                const _Float16* __restrict__ w2h,
                                                const float* __restrict__ bn1s,
                                                const float* __restrict__ bn1b,
                                                const float* __restrict__ bn2s,
                                                const float* __restrict__ bn2b,
                                                const float* __restrict__ gates,
                                                float* __restrict__ out) {
  // union: first 8 KB holds the double-buffered im2col K-slab during stage 1;
  // whole 64 KB is the hidden tile h[n][hid] afterwards (barrier-separated).
  __shared__ _Float16 smem[NT * HID];  // 64 KB
  _Float16* hlds = smem;               // [nn][HID] (stage-1 output / stage-2 B)

  int be = blockIdx.y;
  int b = be >> 3, e = be & 7;
  float gate = gates[b * E_ + e];
  if (gate == 0.0f) return;            // block-uniform, before any barrier

  int n0 = blockIdx.x * NT;
  int tid = threadIdx.x, wave = tid >> 5, lane = tid & 31;
  int g = lane >> 4, ln = lane & 15;
  const _Float16* xb = xh + (size_t)b * C_ * HW;

  // precompute per-thread staging indices: 32k x 64n = 2048 elems, 8 per thread
  int cb_[8], py_[8], px_[8], lo_[8];
#pragma unroll
  for (int i = 0; i < 8; ++i) {
    int q  = tid + i * 256;
    int kk = q >> 6;                   // 0..31 (channel within chunk)
    int nn = q & 63;                   // pixel within tile
    int n  = n0 + nn;
    int py = n / Ww, px = n % Ww;
    py_[i] = py; px_[i] = px;
    cb_[i] = kk * HW + py * Ww + px;
    lo_[i] = nn * 32 + kk;
  }

  // ---- stage 1: h[512][64] = silu(bn1(W1 @ im2col(x))) ----
  v8f acc[4][4] = {};
#pragma unroll 1
  for (int kt = 0; kt < K1 / 32; ++kt) {
    int r  = kt >> 3;                  // 0..8 : kh*3+kw (uniform)
    int c0 = (kt & 7) * 32;
    int dy = r / 3 - 1, dx = r % 3 - 1;
    int doff = c0 * HW + dy * Ww + dx;
    _Float16* Bs = smem + (kt & 1) * (NT * 32);   // ping-pong 4 KB slabs
#pragma unroll
    for (int i = 0; i < 8; ++i) {
      int iy = py_[i] + dy, ix = px_[i] + dx;
      _Float16 v = (_Float16)0.0f;
      if ((unsigned)iy < (unsigned)Hh && (unsigned)ix < (unsigned)Ww)
        v = xb[cb_[i] + doff];
      Bs[lo_[i]] = v;
    }
    __syncthreads();                   // staged slab visible; prev slab free
    int k0 = kt * 32;
    v16h bfrag[4];
#pragma unroll
    for (int nt = 0; nt < 4; ++nt) {
      const _Float16* bp = &Bs[(nt * 16 + ln) * 32 + g * 8];
      bfrag[nt] = join8(*(const v8h*)bp, *(const v8h*)(bp + 16));
    }
#pragma unroll
    for (int mc = 0; mc < 4; ++mc) {
      int m = (wave * 4 + mc) * 16 + ln;
      const _Float16* ap = w1h + ((size_t)(e * HID) + m) * K1 + k0;
      v16h af = join8(*(const v8h*)(ap + g * 8), *(const v8h*)(ap + 16 + g * 8));
#pragma unroll
      for (int nt = 0; nt < 4; ++nt)
        acc[mc][nt] = wmma_f16(af, bfrag[nt], acc[mc][nt]);
    }
  }
  __syncthreads();                     // Bs fully dead before hlds overwrite

  // epilogue 1: BN1 + SiLU -> hlds (f16)
#pragma unroll
  for (int mc = 0; mc < 4; ++mc)
#pragma unroll
    for (int j = 0; j < 8; ++j) {
      int m = (wave * 4 + mc) * 16 + g * 8 + j;
      float inv = bn1s[e * HID + m] * INV_SQRT_1PEPS;
      float bi = bn1b[e * HID + m];
#pragma unroll
      for (int nt = 0; nt < 4; ++nt) {
        float v = acc[mc][nt][j] * inv + bi;
        hlds[(nt * 16 + ln) * HID + m] = (_Float16)silu_f(v);
      }
    }
  __syncthreads();

  // ---- stage 2: y[256][64] = bn2(W2 @ h); atomically add gate*y ----
  v8f acc2[2][4] = {};
#pragma unroll 1
  for (int kt = 0; kt < HID / 32; ++kt) {
    int k0 = kt * 32;
    v16h bfrag[4];
#pragma unroll
    for (int nt = 0; nt < 4; ++nt) {
      const _Float16* bp = &hlds[(nt * 16 + ln) * HID + k0 + g * 8];
      bfrag[nt] = join8(*(const v8h*)bp, *(const v8h*)(bp + 16));
    }
#pragma unroll
    for (int mc = 0; mc < 2; ++mc) {
      int m = (wave + mc * 8) * 16 + ln;
      const _Float16* ap = w2h + ((size_t)(e * O_) + m) * HID + k0;
      v16h af = join8(*(const v8h*)(ap + g * 8), *(const v8h*)(ap + 16 + g * 8));
#pragma unroll
      for (int nt = 0; nt < 4; ++nt)
        acc2[mc][nt] = wmma_f16(af, bfrag[nt], acc2[mc][nt]);
    }
  }
#pragma unroll
  for (int mc = 0; mc < 2; ++mc)
#pragma unroll
    for (int j = 0; j < 8; ++j) {
      int m = (wave + mc * 8) * 16 + g * 8 + j;
      float inv = bn2s[e * O_ + m] * INV_SQRT_1PEPS;
      float bi = bn2b[e * O_ + m];
#pragma unroll
      for (int nt = 0; nt < 4; ++nt) {
        int n = n0 + nt * 16 + ln;
        float v = (acc2[mc][nt][j] * inv + bi) * gate;
        atomicAdd(&out[((size_t)(b * O_ + m)) * HW + n], v);
      }
    }
}

// ---------------- host: workspace layout + launches ----------------
extern "C" void kernel_launch(void* const* d_in, const int* in_sizes, int n_in,
                              void* d_out, int out_size, void* d_ws, size_t ws_size,
                              hipStream_t stream) {
  (void)in_sizes; (void)n_in; (void)out_size; (void)ws_size;
  const float* x        = (const float*)d_in[0];
  const float* router_w = (const float*)d_in[1];
  const float* router_b = (const float*)d_in[2];
  const float* shared_w = (const float*)d_in[3];
  const float* shared_s = (const float*)d_in[4];
  const float* shared_b = (const float*)d_in[5];
  const float* w1       = (const float*)d_in[6];
  const float* bn1s     = (const float*)d_in[7];
  const float* bn1b     = (const float*)d_in[8];
  const float* w2       = (const float*)d_in[9];
  const float* bn2s     = (const float*)d_in[10];
  const float* bn2b     = (const float*)d_in[11];
  float* out = (float*)d_out;

  char* ws = (char*)d_ws;
  constexpr size_t off_gates  = 0;                                  // 128 f32
  constexpr size_t off_pooled = 512;                                // 4096 f32
  constexpr size_t off_xh     = 16896;                              // 16B aligned
  constexpr size_t sz_xh      = (size_t)B_ * C_ * HW * 2;           // 25,690,112
  constexpr size_t off_w1h    = off_xh + sz_xh;
  constexpr size_t sz_w1h     = (size_t)E_ * HID * K1 * 2;          // 18,874,368
  constexpr size_t off_w2h    = off_w1h + sz_w1h;
  constexpr size_t sz_w2h     = (size_t)E_ * O_ * HID * 2;          // 2,097,152
  constexpr size_t off_swh    = off_w2h + sz_w2h;

  float*    gates  = (float*)(ws + off_gates);
  float*    pooled = (float*)(ws + off_pooled);
  _Float16* xh     = (_Float16*)(ws + off_xh);
  _Float16* w1h    = (_Float16*)(ws + off_w1h);
  _Float16* w2h    = (_Float16*)(ws + off_w2h);
  _Float16* swh    = (_Float16*)(ws + off_swh);

  k_f32_to_f16<<<4096, 256, 0, stream>>>(x, xh, B_ * C_ * HW);
  k_convert_w1<<<4096, 256, 0, stream>>>(w1, w1h, E_ * HID * K1);
  k_f32_to_f16<<<1024, 256, 0, stream>>>(w2, w2h, E_ * O_ * HID);
  k_f32_to_f16<<<64,   256, 0, stream>>>(shared_w, swh, O_ * C_);

  k_pool<<<B_ * C_, 256, 0, stream>>>(x, pooled);
  k_router<<<1, 128, 0, stream>>>(pooled, router_w, router_b, gates);

  k_shared<<<dim3(NTILES, B_), 256, 0, stream>>>(xh, swh, shared_s, shared_b, x, out);
  k_expert<<<dim3(NTILES, B_ * E_), 256, 0, stream>>>(xh, w1h, w2h, bn1s, bn1b,
                                                      bn2s, bn2b, gates, out);
}
